// CrystalTransformer_17454747091356
// MI455X (gfx1250) — compile-verified, matrix operand order checked by reading, not run on hardware
//
#include <hip/hip_runtime.h>
#include <stdint.h>

// ---------------------------------------------------------------------------
// Types for CDNA5 WMMA (wave32, 16x16x32 bf16 -> f32)
// ---------------------------------------------------------------------------
typedef __attribute__((ext_vector_type(16))) __bf16 v16bf;
typedef __attribute__((ext_vector_type(8)))  float  v8f;

union FragBF {
    v16bf    v;
    uint32_t u[8];
    uint4    q[2];
};

// pack two fp32 into one dword of two bf16 (round-to-nearest-even), low = first
__device__ inline uint32_t bf16pack(float a, float b) {
    uint32_t ua = __float_as_uint(a), ub = __float_as_uint(b);
    ua = (ua + 0x7FFFu + ((ua >> 16) & 1u)) >> 16;
    ub = (ub + 0x7FFFu + ((ub >> 16) & 1u)) >> 16;
    return (ub << 16) | ua;
}

__device__ inline float rbf_val(float d, int j, float step, float inv) {
    float t = d - step * (float)j;
    return __expf(-t * t * inv);
}

// RBF constants (NRBF=32)
#define ESTEP (5.0f / 31.0f)
#define EINV  40.96f                 // (32/5)^2
#define ASTEP (3.14159265358979323846f / 31.0f)
#define AINV  103.75285653f          // (32/pi)^2
#define DSTEP (10.0f / 31.0f)
#define DINV  10.24f                 // (32/10)^2

// ---------------------------------------------------------------------------
// Core: one block (4 waves, 128 thr) computes C[32x128] = act(A[32xK] @ W + b)
// Each wave owns a 32x32 sub-tile: 2 M-tiles x 2 N-tiles = 4 WMMA per k-step,
// so each B fragment (2x global_load_b128) feeds 2 WMMAs and each A fragment
// (2x ds_load_b128) feeds 2 WMMAs.
// A: bf16 pair-packed rows in LDS (row stride = KP dwords, KP = K/2); pair
//    index within a k-tile is p = (i/4)*8 + half*4 + (i%4) per the CDNA5
//    16-bit A (16x32) VGPR layout -> each 4-group consecutive => ds_load_b128.
// W: FRAGMENT-MAJOR pack in global: lane l of fragment (kt, nt) holds 8
//    contiguous dwords at ((kt*NT + nt)*32 + l)*8 => two global_load_b128.
// ---------------------------------------------------------------------------
__device__ inline void wmma_block_32x128(const uint32_t* __restrict__ Ap, int KP,
                                         const uint32_t* __restrict__ Wp, int NT,
                                         int ntbase,
                                         const float* __restrict__ bias, int act,
                                         float* __restrict__ C) {
    const int lane = threadIdx.x & 31;
    const int wid  = threadIdx.x >> 5;      // 0..3, owns cols [wid*32, wid*32+32)
    const int half = lane >> 4;
    const int mn   = lane & 15;             // A row / B-C column within tile
    const int nt0  = ntbase + wid * 2;      // global 16-col tile index
    v8f acc00 = {}, acc01 = {}, acc10 = {}, acc11 = {};
    const int nkt = KP >> 4;                // K/32 tiles
    for (int kt = 0; kt < nkt; ++kt) {
        FragBF a0, a1, b0, b1;
        const uint32_t* arow0 = Ap + mn * KP + (kt << 4);
        const uint32_t* arow1 = arow0 + 16 * KP;
#pragma unroll
        for (int i = 0; i < 8; ++i) {
            int p = ((i >> 2) << 3) + (half << 2) + (i & 3);
            a0.u[i] = arow0[p];
            a1.u[i] = arow1[p];
        }
        const uint4* s0 = (const uint4*)(Wp + (((size_t)(kt * NT + nt0)) * 32 + lane) * 8);
        const uint4* s1 = (const uint4*)(Wp + (((size_t)(kt * NT + nt0 + 1)) * 32 + lane) * 8);
        b0.q[0] = s0[0]; b0.q[1] = s0[1];
        b1.q[0] = s1[0]; b1.q[1] = s1[1];
        acc00 = __builtin_amdgcn_wmma_f32_16x16x32_bf16(false, a0.v, false, b0.v,
                                                        (short)0, acc00, false, false);
        acc01 = __builtin_amdgcn_wmma_f32_16x16x32_bf16(false, a0.v, false, b1.v,
                                                        (short)0, acc01, false, false);
        acc10 = __builtin_amdgcn_wmma_f32_16x16x32_bf16(false, a1.v, false, b0.v,
                                                        (short)0, acc10, false, false);
        acc11 = __builtin_amdgcn_wmma_f32_16x16x32_bf16(false, a1.v, false, b1.v,
                                                        (short)0, acc11, false, false);
    }
    // C/D layout: element r -> row (half*8 + r), col = lane%16 (+16 for *1 accs)
    float bn0 = bias ? bias[nt0 * 16 + mn]       : 0.f;
    float bn1 = bias ? bias[(nt0 + 1) * 16 + mn] : 0.f;
#pragma unroll
    for (int r = 0; r < 8; ++r) {
        int row0 = (half << 3) + r;
        float v00 = acc00[r] + bn0, v01 = acc01[r] + bn1;
        float v10 = acc10[r] + bn0, v11 = acc11[r] + bn1;
        if (act == 1) {                     // SiLU
            v00 = v00 / (1.f + __expf(-v00));
            v01 = v01 / (1.f + __expf(-v01));
            v10 = v10 / (1.f + __expf(-v10));
            v11 = v11 / (1.f + __expf(-v11));
        }
        C[row0 * 128 + wid * 32 + mn]             = v00;
        C[row0 * 128 + wid * 32 + 16 + mn]        = v01;
        C[(16 + row0) * 128 + wid * 32 + mn]      = v10;
        C[(16 + row0) * 128 + wid * 32 + 16 + mn] = v11;
    }
}

// ---------------------------------------------------------------------------
// Weight pre-pack: fp32 (K x N) row-major -> fragment-major bf16 pair dwords.
// Linear index = ((kt*NT + nt)*32 + lane)*8 + i.
// ---------------------------------------------------------------------------
__global__ void pack_frag_kernel(const float* __restrict__ W, uint32_t* __restrict__ P,
                                 int K, int N) {
    int idx = blockIdx.x * blockDim.x + threadIdx.x;
    int total = (K >> 1) * N;
    if (idx >= total) return;
    int i    = idx & 7;
    int lane = (idx >> 3) & 31;
    int f    = idx >> 8;                    // kt*NT + nt
    int NT   = N >> 4;
    int kt = f / NT, nt = f % NT;
    int half = lane >> 4;
    int n    = nt * 16 + (lane & 15);
    int p    = kt * 16 + ((i >> 2) << 3) + (half << 2) + (i & 3);
    P[idx] = bf16pack(W[(size_t)(2 * p) * N + n], W[(size_t)(2 * p + 1) * N + n]);
}

// ---------------------------------------------------------------------------
// Generic GEMM: Out[32x128 tile] = act(A @ W + b); outmode 0=store 1=add
// ---------------------------------------------------------------------------
template <int KP>
__global__ void gemm_tile_kernel(const float* __restrict__ A, int lda,
                                 const uint32_t* __restrict__ Wp, int NT,
                                 const float* __restrict__ bias,
                                 float* __restrict__ Out, int ldo,
                                 int act, int outmode) {
    __shared__ uint32_t Ap[32 * KP];
    __shared__ float    C[32 * 128];
    const int row0 = blockIdx.x * 32;
    const int colbase = blockIdx.y * 128;
    for (int idx = threadIdx.x; idx < 32 * KP; idx += 128) {
        int r = idx / KP, p = idx % KP;
        const float* ar = A + (size_t)(row0 + r) * lda + 2 * p;
        Ap[idx] = bf16pack(ar[0], ar[1]);
    }
    __syncthreads();
    wmma_block_32x128(Ap, KP, Wp, NT, blockIdx.y * 8, bias, act, C);
    __syncthreads();
    for (int idx = threadIdx.x; idx < 4096; idx += 128) {
        int r = idx >> 7, c = idx & 127;
        float v = C[idx];
        float* o = Out + (size_t)(row0 + r) * ldo + colbase + c;
        if (outmode == 0) *o = v; else *o += v;
    }
}

// ---------------------------------------------------------------------------
// Edge MLP: in = [flat[row] | flat[col] | rbf(edge_dist)] (K=288), 2 layers,
// scatter-add result into aggr[row]. 32 edges per block.
// ---------------------------------------------------------------------------
__global__ void edge_mlp_kernel(const float* __restrict__ flat,
                                const int* __restrict__ eidx, int E,
                                const float* __restrict__ edist,
                                const uint32_t* __restrict__ W1p, const float* __restrict__ b1,
                                const uint32_t* __restrict__ W2p, const float* __restrict__ b2,
                                float* __restrict__ aggr) {
    __shared__ uint32_t Ap[32 * 144];
    __shared__ float    C[32 * 128];
    __shared__ uint32_t Hp[32 * 64];
    __shared__ int      rowidx[32];
    const int e0 = blockIdx.x * 32;
    if (threadIdx.x < 32) rowidx[threadIdx.x] = eidx[e0 + threadIdx.x];
    for (int idx = threadIdx.x; idx < 32 * 144; idx += 128) {
        int r = idx / 144, p = idx % 144;
        int e = e0 + r;
        float a0, a1;
        if (p < 64) {
            int s = eidx[e];
            a0 = flat[(size_t)s * 128 + 2 * p]; a1 = flat[(size_t)s * 128 + 2 * p + 1];
        } else if (p < 128) {
            int s = eidx[E + e]; int k = 2 * (p - 64);
            a0 = flat[(size_t)s * 128 + k]; a1 = flat[(size_t)s * 128 + k + 1];
        } else {
            int j = 2 * (p - 128); float d = edist[e];
            a0 = rbf_val(d, j, ESTEP, EINV); a1 = rbf_val(d, j + 1, ESTEP, EINV);
        }
        Ap[r * 144 + p] = bf16pack(a0, a1);
    }
    __syncthreads();
    wmma_block_32x128(Ap, 144, W1p, 8, 0, b1, 1, C);
    __syncthreads();
    for (int idx = threadIdx.x; idx < 32 * 64; idx += 128) {
        int r = idx >> 6, p = idx & 63;
        Hp[idx] = bf16pack(C[r * 128 + 2 * p], C[r * 128 + 2 * p + 1]);
    }
    __syncthreads();
    wmma_block_32x128(Hp, 64, W2p, 8, 0, b2, 0, C);
    __syncthreads();
    for (int idx = threadIdx.x; idx < 4096; idx += 128) {
        int r = idx >> 7, c = idx & 127;
        atomicAdd(&aggr[(size_t)rowidx[r] * 128 + c], C[idx]);
    }
}

// ---------------------------------------------------------------------------
// Triplet MLP: in = [flat[center] | rbf(angle)] (K=160), scatter to aggr[center]
// ---------------------------------------------------------------------------
__global__ void trip_mlp_kernel(const float* __restrict__ flat,
                                const int* __restrict__ tidx,
                                const float* __restrict__ ang,
                                const uint32_t* __restrict__ W1p, const float* __restrict__ b1,
                                const uint32_t* __restrict__ W2p, const float* __restrict__ b2,
                                float* __restrict__ aggr) {
    __shared__ uint32_t Ap[32 * 80];
    __shared__ float    C[32 * 128];
    __shared__ uint32_t Hp[32 * 64];
    __shared__ int      cidx[32];
    const int e0 = blockIdx.x * 32;
    if (threadIdx.x < 32) cidx[threadIdx.x] = tidx[(size_t)(e0 + threadIdx.x) * 3 + 1];
    __syncthreads();
    for (int idx = threadIdx.x; idx < 32 * 80; idx += 128) {
        int r = idx / 80, p = idx % 80;
        int e = e0 + r;
        float a0, a1;
        if (p < 64) {
            int s = cidx[r];
            a0 = flat[(size_t)s * 128 + 2 * p]; a1 = flat[(size_t)s * 128 + 2 * p + 1];
        } else {
            int j = 2 * (p - 64); float d = ang[e];
            a0 = rbf_val(d, j, ASTEP, AINV); a1 = rbf_val(d, j + 1, ASTEP, AINV);
        }
        Ap[r * 80 + p] = bf16pack(a0, a1);
    }
    __syncthreads();
    wmma_block_32x128(Ap, 80, W1p, 8, 0, b1, 1, C);
    __syncthreads();
    for (int idx = threadIdx.x; idx < 32 * 64; idx += 128) {
        int r = idx >> 6, p = idx & 63;
        Hp[idx] = bf16pack(C[r * 128 + 2 * p], C[r * 128 + 2 * p + 1]);
    }
    __syncthreads();
    wmma_block_32x128(Hp, 64, W2p, 8, 0, b2, 0, C);
    __syncthreads();
    for (int idx = threadIdx.x; idx < 4096; idx += 128) {
        int r = idx >> 7, c = idx & 127;
        atomicAdd(&aggr[(size_t)cidx[r] * 128 + c], C[idx]);
    }
}

// ---------------------------------------------------------------------------
// Node MLP: in = [flat[g] | aggr[g]] (K=256); flat[g] += out (per-row, safe)
// ---------------------------------------------------------------------------
__global__ void node_mlp_kernel(float* __restrict__ flat,
                                const float* __restrict__ aggr,
                                const uint32_t* __restrict__ W1p, const float* __restrict__ b1,
                                const uint32_t* __restrict__ W2p, const float* __restrict__ b2) {
    __shared__ uint32_t Ap[32 * 128];
    __shared__ float    C[32 * 128];
    __shared__ uint32_t Hp[32 * 64];
    const int g0 = blockIdx.x * 32;
    for (int idx = threadIdx.x; idx < 32 * 128; idx += 128) {
        int r = idx >> 7, p = idx & 127;
        int g = g0 + r;
        float a0, a1;
        if (p < 64) {
            a0 = flat[(size_t)g * 128 + 2 * p]; a1 = flat[(size_t)g * 128 + 2 * p + 1];
        } else {
            int k = 2 * (p - 64);
            a0 = aggr[(size_t)g * 128 + k]; a1 = aggr[(size_t)g * 128 + k + 1];
        }
        Ap[idx] = bf16pack(a0, a1);
    }
    __syncthreads();
    wmma_block_32x128(Ap, 128, W1p, 8, 0, b1, 1, C);
    __syncthreads();
    for (int idx = threadIdx.x; idx < 32 * 64; idx += 128) {
        int r = idx >> 6, p = idx & 63;
        Hp[idx] = bf16pack(C[r * 128 + 2 * p], C[r * 128 + 2 * p + 1]);
    }
    __syncthreads();
    wmma_block_32x128(Hp, 64, W2p, 8, 0, b2, 0, C);
    __syncthreads();
    for (int idx = threadIdx.x; idx < 4096; idx += 128) {
        int r = idx >> 7, c = idx & 127;
        flat[(size_t)(g0 + r) * 128 + c] += C[idx];
    }
}

// ---------------------------------------------------------------------------
// Bias MLP on distance RBF: rows = B*N*N; l1 via WMMA (K=32), l2 (128->4) VALU
// 32 rows per block -> 32 rows x 4 heads = 128 VALU dot products for l2.
// ---------------------------------------------------------------------------
__global__ void biasmlp_kernel(const float* __restrict__ dist,
                               const uint32_t* __restrict__ W1p, const float* __restrict__ b1,
                               const float* __restrict__ W2, const float* __restrict__ b2,
                               float* __restrict__ biasbuf) {
    __shared__ uint32_t Ap[32 * 16];
    __shared__ float    C[32 * 128];
    const int e0 = blockIdx.x * 32;
    for (int idx = threadIdx.x; idx < 32 * 16; idx += 128) {
        int r = idx >> 4, p = idx & 15;
        float d = dist[e0 + r];
        Ap[idx] = bf16pack(rbf_val(d, 2 * p, DSTEP, DINV),
                           rbf_val(d, 2 * p + 1, DSTEP, DINV));
    }
    __syncthreads();
    wmma_block_32x128(Ap, 16, W1p, 8, 0, b1, 1, C);
    __syncthreads();
    {
        int r = threadIdx.x >> 2, hh = threadIdx.x & 3;
        float acc = b2[hh];
        for (int c = 0; c < 128; ++c) acc += C[r * 128 + c] * W2[c * 4 + hh];
        int e = e0 + r;
        int j = e & 127, i = (e >> 7) & 127, bb = e >> 14;
        biasbuf[(((size_t)bb * 4 + hh) * 128 + i) * 128 + j] = acc;
    }
}

// ---------------------------------------------------------------------------
// Attention, single pass online softmax: block = (b, head), thread = query row
// ---------------------------------------------------------------------------
__global__ void attn_kernel(const float* __restrict__ qkv,
                            const float* __restrict__ biasbuf,
                            const unsigned char* __restrict__ mask,
                            float* __restrict__ out) {
    __shared__ float Ks[128][33];
    __shared__ float Vs[128][33];
    const int bb = blockIdx.x >> 2, hh = blockIdx.x & 3;
    const int base = bb * 128;
    for (int idx = threadIdx.x; idx < 128 * 32; idx += 128) {
        int m = idx >> 5, d = idx & 31;
        const float* row = qkv + (size_t)(base + m) * 384 + hh * 32 + d;
        Ks[m][d] = row[128];
        Vs[m][d] = row[256];
    }
    __syncthreads();
    const int i = threadIdx.x;
    float q[32];
    const float* qrow = qkv + (size_t)(base + i) * 384 + hh * 32;
#pragma unroll
    for (int d = 0; d < 32; ++d) q[d] = qrow[d];
    const float* brow = biasbuf + (((size_t)bb * 4 + hh) * 128 + i) * 128;
    float mrun = -3.0e38f, l = 0.f, acc[32];
#pragma unroll
    for (int d = 0; d < 32; ++d) acc[d] = 0.f;
    const float scale = 0.17677669529663687f;   // 1/sqrt(32)
    for (int m = 0; m < 128; ++m) {
        float s = 0.f;
#pragma unroll
        for (int d = 0; d < 32; ++d) s += q[d] * Ks[m][d];
        s = s * scale + brow[m];
        if (!mask[base + m]) s = -1.0e9f;
        float mnew = fmaxf(mrun, s);
        float cs = __expf(mrun - mnew);
        float p  = __expf(s - mnew);
        l = l * cs + p;
#pragma unroll
        for (int d = 0; d < 32; ++d) acc[d] = acc[d] * cs + p * Vs[m][d];
        mrun = mnew;
    }
    float invl = 1.f / l;
    float* orow = out + (size_t)(base + i) * 128 + hh * 32;
#pragma unroll
    for (int d = 0; d < 32; ++d) orow[d] = acc[d] * invl;
}

// ---------------------------------------------------------------------------
// LayerNorm over 128 features: one block per row
// ---------------------------------------------------------------------------
__global__ void ln_kernel(const float* __restrict__ in,
                          const float* __restrict__ g, const float* __restrict__ b,
                          float* __restrict__ out) {
    __shared__ float red[128];
    const int row = blockIdx.x, t = threadIdx.x;
    float v = in[(size_t)row * 128 + t];
    red[t] = v; __syncthreads();
    for (int s = 64; s > 0; s >>= 1) { if (t < s) red[t] += red[t + s]; __syncthreads(); }
    float mean = red[0] / 128.f; __syncthreads();
    float dv = v - mean;
    red[t] = dv * dv; __syncthreads();
    for (int s = 64; s > 0; s >>= 1) { if (t < s) red[t] += red[t + s]; __syncthreads(); }
    float var = red[0] / 128.f;
    out[(size_t)row * 128 + t] = dv * rsqrtf(var + 1e-5f) * g[t] + b[t];
}

// ---------------------------------------------------------------------------
// Embedding (K=9): one thread per output element
// ---------------------------------------------------------------------------
__global__ void embed_kernel(const float* __restrict__ x,
                             const float* __restrict__ W, const float* __restrict__ b,
                             float* __restrict__ flat) {
    int idx = blockIdx.x * blockDim.x + threadIdx.x;
    int g = idx >> 7, n = idx & 127;
    const float* xr = x + (size_t)g * 9;
    float acc = b[n];
#pragma unroll
    for (int k = 0; k < 9; ++k) acc += xr[k] * W[k * 128 + n];
    flat[idx] = acc;
}

// ---------------------------------------------------------------------------
// Readout: masked mean over N, then 128->128 SiLU -> 1
// ---------------------------------------------------------------------------
__global__ void readout_kernel(const float* __restrict__ flat,
                               const unsigned char* __restrict__ mask,
                               const float* __restrict__ W1, const float* __restrict__ b1,
                               const float* __restrict__ W2, const float* __restrict__ b2,
                               float* __restrict__ out) {
    __shared__ float gf[128];
    __shared__ float hid[128];
    const int bb = blockIdx.x, t = threadIdx.x;
    float s = 0.f, cnt = 0.f;
    for (int i = 0; i < 128; ++i) {
        float mm = mask[bb * 128 + i] ? 1.f : 0.f;
        s += mm * flat[((size_t)bb * 128 + i) * 128 + t];
        cnt += mm;
    }
    gf[t] = s / fmaxf(cnt, 1e-9f);
    __syncthreads();
    float acc = b1[t];
    for (int c = 0; c < 128; ++c) acc += gf[c] * W1[c * 128 + t];
    acc = acc / (1.f + __expf(-acc));
    hid[t] = acc * W2[t];
    __syncthreads();
    for (int sft = 64; sft > 0; sft >>= 1) { if (t < sft) hid[t] += hid[t + sft]; __syncthreads(); }
    if (t == 0) out[bb] = hid[0] + b2[0];
}

// ---------------------------------------------------------------------------
// Host orchestration
// ---------------------------------------------------------------------------
// Pytree-flattened input order (dicts in sorted key order):
//  0:x 1:dist_matrix 2:edge_dist 3:angles
//  4:embed.b 5:embed.w
//  global[i] base Gi = 6 + 16*i:
//   +0 bias.l1.b +1 bias.l1.w +2 bias.l2.b +3 bias.l2.w +4 ffn.l1.b +5 ffn.l1.w
//   +6 ffn.l2.b  +7 ffn.l2.w  +8 ln1.b +9 ln1.g +10 ln2.b +11 ln2.g
//   +12 out.b +13 out.w +14 qkv.b +15 qkv.w
//  local[i] base Li = 38 + 12*i:
//   +0 edge.l1.b +1 edge.l1.w +2 edge.l2.b +3 edge.l2.w
//   +4 node.l1.b +5 node.l1.w +6 node.l2.b +7 node.l2.w
//   +8 trip.l1.b +9 trip.l1.w +10 trip.l2.b +11 trip.l2.w
//  74:readout.l1.b 75:readout.l1.w 76:readout.l2.b 77:readout.l2.w
//  78:atom_mask 79:edge_index 80:triplet_index
extern "C" void kernel_launch(void* const* d_in, const int* in_sizes, int n_in,
                              void* d_out, int out_size, void* d_ws, size_t ws_size,
                              hipStream_t stream) {
    (void)in_sizes; (void)n_in; (void)out_size; (void)ws_size;
    const int Bn = 4096, E = 98304, M = 262144;

    auto F = [&](int i) -> const float* { return (const float*)d_in[i]; };
    const float* x     = F(0);
    const float* dist  = F(1);
    const float* edist = F(2);
    const float* ang   = F(3);
    const unsigned char* mask = (const unsigned char*)d_in[78];
    const int* eidx = (const int*)d_in[79];
    const int* tidx = (const int*)d_in[80];

    char* wsb = (char*)d_ws;
    size_t off = 0;
    auto alloc = [&](size_t bytes) -> char* {
        char* p = wsb + off;
        off = (off + bytes + 255) & ~(size_t)255;
        return p;
    };
    float* flat    = (float*)alloc((size_t)Bn * 128 * 4);
    float* aggr    = (float*)alloc((size_t)Bn * 128 * 4);
    float* xn      = (float*)alloc((size_t)Bn * 128 * 4);
    float* qkvbuf  = (float*)alloc((size_t)Bn * 384 * 4);
    float* attnout = (float*)alloc((size_t)Bn * 128 * 4);
    float* hid     = (float*)alloc((size_t)Bn * 512 * 4);
    float* biasbuf = (float*)alloc((size_t)32 * 4 * 128 * 128 * 4);

    auto packW = [&](int pidx, int K, int N) -> const uint32_t* {
        uint32_t* P = (uint32_t*)alloc((size_t)(K / 2) * N * 4);
        int total = (K / 2) * N;
        pack_frag_kernel<<<(total + 255) / 256, 256, 0, stream>>>(F(pidx), P, K, N);
        return P;
    };

    const uint32_t *Pqkv[2], *Pout[2], *Pffn1[2], *Pffn2[2], *Pbias1[2];
    for (int i = 0; i < 2; ++i) {
        int b = 6 + 16 * i;
        Pbias1[i] = packW(b + 1, 32, 128);
        Pffn1[i]  = packW(b + 5, 128, 512);
        Pffn2[i]  = packW(b + 7, 512, 128);
        Pout[i]   = packW(b + 13, 128, 128);
        Pqkv[i]   = packW(b + 15, 128, 384);
    }
    const uint32_t *Pe1[3], *Pe2[3], *Pn1[3], *Pn2[3], *Pt1[3], *Pt2[3];
    for (int i = 0; i < 3; ++i) {
        int b = 38 + 12 * i;
        Pe1[i] = packW(b + 1, 288, 128);
        Pe2[i] = packW(b + 3, 128, 128);
        Pn1[i] = packW(b + 5, 256, 128);
        Pn2[i] = packW(b + 7, 128, 128);
        Pt1[i] = packW(b + 9, 160, 128);
        Pt2[i] = packW(b + 11, 128, 128);
    }

    // ---- embed ----
    embed_kernel<<<(Bn * 128) / 256, 256, 0, stream>>>(x, F(5), F(4), flat);

    // ---- 3 local message-passing layers ----
    for (int i = 0; i < 3; ++i) {
        int b = 38 + 12 * i;
        hipMemsetAsync(aggr, 0, (size_t)Bn * 128 * 4, stream);
        edge_mlp_kernel<<<E / 32, 128, 0, stream>>>(flat, eidx, E, edist,
                                                    Pe1[i], F(b + 0), Pe2[i], F(b + 2), aggr);
        trip_mlp_kernel<<<M / 32, 128, 0, stream>>>(flat, tidx, ang,
                                                    Pt1[i], F(b + 8), Pt2[i], F(b + 10), aggr);
        node_mlp_kernel<<<Bn / 32, 128, 0, stream>>>(flat, aggr,
                                                     Pn1[i], F(b + 4), Pn2[i], F(b + 6));
    }

    // ---- 2 global transformer layers ----
    for (int i = 0; i < 2; ++i) {
        int b = 6 + 16 * i;
        ln_kernel<<<Bn, 128, 0, stream>>>(flat, F(b + 9), F(b + 8), xn);              // ln1
        gemm_tile_kernel<64><<<dim3(Bn / 32, 3), 128, 0, stream>>>(
            xn, 128, Pqkv[i], 24, F(b + 14), qkvbuf, 384, 0, 0);                       // qkv
        biasmlp_kernel<<<(32 * 128 * 128) / 32, 128, 0, stream>>>(
            dist, Pbias1[i], F(b + 0), F(b + 3), F(b + 2), biasbuf);                   // attn bias
        attn_kernel<<<32 * 4, 128, 0, stream>>>(qkvbuf, biasbuf, mask, attnout);
        gemm_tile_kernel<64><<<dim3(Bn / 32, 1), 128, 0, stream>>>(
            attnout, 128, Pout[i], 8, F(b + 12), flat, 128, 0, 1);                     // proj + residual
        ln_kernel<<<Bn, 128, 0, stream>>>(flat, F(b + 11), F(b + 10), xn);            // ln2
        gemm_tile_kernel<64><<<dim3(Bn / 32, 4), 128, 0, stream>>>(
            xn, 128, Pffn1[i], 32, F(b + 4), hid, 512, 1, 0);                          // ffn l1 + SiLU
        gemm_tile_kernel<256><<<dim3(Bn / 32, 1), 128, 0, stream>>>(
            hid, 512, Pffn2[i], 8, F(b + 6), flat, 128, 0, 1);                         // ffn l2 + residual
    }

    // ---- readout ----
    readout_kernel<<<32, 128, 0, stream>>>(flat, mask, F(75), F(74), F(77), F(76),
                                           (float*)d_out);
}
